// MultiTaskMoE_25357486915858
// MI455X (gfx1250) — matrix operand order repaired and checked by reading, not compiled, not run on hardware
//
#include <hip/hip_runtime.h>
#include <hip/hip_bf16.h>

// ---------------------------------------------------------------------------
// MultiTaskMoE forward for gfx1250 (MI455X).
//  - Dense GEMMs: one wave per 16x64 tile, K templated (128/256) for full
//    unroll -> dense v_wmma_f32_16x16x32_f16 issue with A-fragment reuse.
//  - N^2 attention (dist-bias MLP + masked online softmax + ctx + sum attn*r)
//    fused into one wave per (row, head) with wave32 shuffle reductions.
// ---------------------------------------------------------------------------

typedef __attribute__((ext_vector_type(16))) _Float16 v16h;
typedef __attribute__((ext_vector_type(8)))  float    v8f;

static constexpr int   NN    = 512;
static constexpr int   CC    = 128;
static constexpr int   HH    = 4;
static constexpr int   HD    = 32;   // CC / HH
static constexpr int   NELEM = 95;
static constexpr float ATTN_SCALE = 0.17677669529663687f;  // HD^-0.5

__device__ __forceinline__ float gelu_f(float x) {
  const float c0 = 0.7978845608028654f;  // sqrt(2/pi)
  float x3 = x * x * x;
  return 0.5f * x * (1.0f + tanhf(c0 * (x + 0.044715f * x3)));
}

__device__ __forceinline__ float act_apply(float x, int act) {
  if (act == 1) return fmaxf(x, 0.0f);
  if (act == 2) return gelu_f(x);
  return x;
}

// ---------------------------------------------------------------------------
// GEMM: out[N, Dout] = act(A[N,K] @ W[Dout,K]^T + bias) (+ residual)
// Wave computes a 16x64 tile: A fragment loaded once per K-step, reused for
// 4 WMMAs. K is compile-time -> fully unrolled WMMA stream.
// ---------------------------------------------------------------------------
template <int K>
__global__ void gemm_wmma_kernel(const float* __restrict__ A,
                                 const float* __restrict__ W,
                                 const float* __restrict__ bias,
                                 const float* __restrict__ resid,
                                 float* __restrict__ out,
                                 int Dout, int act) {
  int lane = threadIdx.x & 31;
  int m0   = (blockIdx.x * blockDim.y + threadIdx.y) * 16;   // 4 M-tiles/block
  int n0   = blockIdx.y * 64;
  int half = lane >> 4;   // 0 or 1
  int l16  = lane & 15;

  v8f acc[4] = {{}, {}, {}, {}};
#pragma unroll
  for (int kk = 0; kk < K; kk += 32) {
    // A fragment: lane holds row (m0+l16); K indices contiguous, half*16 + e (same as B).
    const float* arow = A + (size_t)(m0 + l16) * K + kk;
    v16h af;
#pragma unroll
    for (int e = 0; e < 16; ++e) {
      // K = half*16 + e (contiguous, matches B)
      af[e] = (_Float16)arow[half * 16 + e];
    }
    // 4 B fragments: lane holds W row (n0+nt*16+l16) == B column; K = half*16+e.
#pragma unroll
    for (int nt = 0; nt < 4; ++nt) {
      const float* wrow = W + (size_t)(n0 + nt * 16 + l16) * K + kk + half * 16;
      v16h bf;
#pragma unroll
      for (int e = 0; e < 16; ++e) bf[e] = (_Float16)wrow[e];
      acc[nt] = __builtin_amdgcn_wmma_f32_16x16x32_f16(
          /*neg_a=*/false, af, /*neg_b=*/false, bf,
          /*c_mod=*/(short)0, acc[nt], /*reuse_a=*/false, /*reuse_b=*/false);
    }
  }

#pragma unroll
  for (int nt = 0; nt < 4; ++nt) {
    float bj = bias[n0 + nt * 16 + l16];
#pragma unroll
    for (int v = 0; v < 8; ++v) {
      int m = m0 + half * 8 + v;                    // D layout: M = 8*half + vgpr
      float val = acc[nt][v] + bj;
      val = act_apply(val, act);
      size_t o = (size_t)m * Dout + n0 + nt * 16 + l16;
      if (resid) val += resid[o];
      out[o] = val;
    }
  }
}

// ---------------------------------------------------------------------------
// Small-output linear: out[n,j] = act(A[n,:]·W[j,:] + b[j]); thread per elem.
// ---------------------------------------------------------------------------
__global__ void linear_small_kernel(const float* __restrict__ A,
                                    const float* __restrict__ W,
                                    const float* __restrict__ bias,
                                    float* __restrict__ out,
                                    int K, int Dout, int act, int nrows) {
  int idx = blockIdx.x * blockDim.x + threadIdx.x;
  if (idx >= nrows * Dout) return;
  int n = idx / Dout, j = idx % Dout;
  const float* a = A + (size_t)n * K;
  const float* w = W + (size_t)j * K;
  float s = bias[j];
  for (int k = 0; k < K; ++k) s += a[k] * w[k];
  out[idx] = act_apply(s, act);
}

// ---------------------------------------------------------------------------
// LayerNorm over C=128: one wave per row (4 elems/lane + shuffle reduce).
// ---------------------------------------------------------------------------
__global__ void layernorm_kernel(const float* __restrict__ x,
                                 const float* __restrict__ g,
                                 const float* __restrict__ b,
                                 float* __restrict__ out) {
  int row  = blockIdx.x * blockDim.y + threadIdx.y;
  int lane = threadIdx.x;
  const float* xr = x + (size_t)row * CC;
  float v[4];
  float s = 0.f;
#pragma unroll
  for (int t = 0; t < 4; ++t) { v[t] = xr[lane + 32 * t]; s += v[t]; }
#pragma unroll
  for (int off = 16; off > 0; off >>= 1) s += __shfl_xor(s, off, 32);
  float mean = s * (1.0f / CC);
  float q = 0.f;
#pragma unroll
  for (int t = 0; t < 4; ++t) { float d = v[t] - mean; q += d * d; }
#pragma unroll
  for (int off = 16; off > 0; off >>= 1) q += __shfl_xor(q, off, 32);
  float rstd = rsqrtf(q * (1.0f / CC) + 1e-5f);
#pragma unroll
  for (int t = 0; t < 4; ++t) {
    int c = lane + 32 * t;
    out[(size_t)row * CC + c] = (v[t] - mean) * rstd * g[c] + b[c];
  }
}

// ---------------------------------------------------------------------------
// Fused attention: one wave per (row i, head h).
// logits = q·k * scale + d2(gelu(d1(dist2))), masked by batch,
// online softmax, ctx[i,h,:] and s_r[i,h,:] = sum_j p_j * r_j.
// ---------------------------------------------------------------------------
__global__ void attention_kernel(const float* __restrict__ q,
                                 const float* __restrict__ k,
                                 const float* __restrict__ v,
                                 const float* __restrict__ r,
                                 const int*   __restrict__ batch,
                                 const float* __restrict__ w1,  // d1.w (C)
                                 const float* __restrict__ b1,  // d1.b (C)
                                 const float* __restrict__ w2,  // d2.w (H,C)
                                 const float* __restrict__ b2,  // d2.b (H)
                                 float* __restrict__ ctx,
                                 float* __restrict__ s_r) {
  int i    = blockIdx.x;
  int h    = threadIdx.y;       // uniform per wave (block = 32 x HH)
  int lane = threadIdx.x;

  float qv[HD];
#pragma unroll
  for (int d = 0; d < HD; ++d) qv[d] = q[(size_t)i * CC + h * HD + d];
  float ri0 = r[i * 3 + 0], ri1 = r[i * 3 + 1], ri2 = r[i * 3 + 2];
  int bi = batch[i];

  const float VERY_NEG = -1.7e38f;
  float m = VERY_NEG, s = 0.f;
  float cacc[HD];
#pragma unroll
  for (int d = 0; d < HD; ++d) cacc[d] = 0.f;
  float racc0 = 0.f, racc1 = 0.f, racc2 = 0.f;

  for (int jb = 0; jb < NN; jb += 32) {
    int j = jb + lane;
    float rj0 = r[j * 3 + 0], rj1 = r[j * 3 + 1], rj2 = r[j * 3 + 2];
    float dx = ri0 - rj0, dy = ri1 - rj1, dz = ri2 - rj2;
    float dist2 = fmaxf(dx * dx + dy * dy + dz * dz, 1e-8f);

    float bias = b2[h];
    for (int c = 0; c < CC; ++c)
      bias += w2[h * CC + c] * gelu_f(w1[c] * dist2 + b1[c]);

    float dot = 0.f;
#pragma unroll
    for (int d = 0; d < HD; ++d) dot += qv[d] * k[(size_t)j * CC + h * HD + d];

    float logit = (batch[j] == bi) ? (dot * ATTN_SCALE + bias) : VERY_NEG;

    float newm = fmaxf(m, logit);
    float so = __expf(m - newm);
    float p  = __expf(logit - newm);
    s = s * so + p;
#pragma unroll
    for (int d = 0; d < HD; ++d)
      cacc[d] = cacc[d] * so + p * v[(size_t)j * CC + h * HD + d];
    racc0 = racc0 * so + p * rj0;
    racc1 = racc1 * so + p * rj1;
    racc2 = racc2 * so + p * rj2;
    m = newm;
  }

  // Cross-lane merge of online-softmax state (butterfly).
#pragma unroll
  for (int off = 16; off > 0; off >>= 1) {
    float m2 = __shfl_xor(m, off, 32);
    float s2 = __shfl_xor(s, off, 32);
    float newm = fmaxf(m, m2);
    float f1 = __expf(m - newm), f2 = __expf(m2 - newm);
#pragma unroll
    for (int d = 0; d < HD; ++d) {
      float t = __shfl_xor(cacc[d], off, 32);
      cacc[d] = cacc[d] * f1 + t * f2;
    }
    float t0 = __shfl_xor(racc0, off, 32);
    float t1 = __shfl_xor(racc1, off, 32);
    float t2 = __shfl_xor(racc2, off, 32);
    racc0 = racc0 * f1 + t0 * f2;
    racc1 = racc1 * f1 + t1 * f2;
    racc2 = racc2 * f1 + t2 * f2;
    s = s * f1 + s2 * f2;
    m = newm;
  }

  float inv = 1.0f / s;
  if (lane == 0) {
#pragma unroll
    for (int d = 0; d < HD; ++d)
      ctx[(size_t)i * CC + h * HD + d] = cacc[d] * inv;
    s_r[((size_t)i * HH + h) * 3 + 0] = racc0 * inv;
    s_r[((size_t)i * HH + h) * 3 + 1] = racc1 * inv;
    s_r[((size_t)i * HH + h) * 3 + 2] = racc2 * inv;
  }
}

// ---------------------------------------------------------------------------
// Coordinate update: gate = clip(sigmoid(g2·g1out + b2)); r += gate*(r - rmean)
// One wave per row.
// ---------------------------------------------------------------------------
__global__ void r_update_kernel(const float* __restrict__ g1o,
                                const float* __restrict__ w2,
                                const float* __restrict__ b2,
                                const float* __restrict__ s_r,
                                float* __restrict__ r) {
  int row  = blockIdx.x * blockDim.y + threadIdx.y;
  int lane = threadIdx.x;
  float s = 0.f;
#pragma unroll
  for (int t = 0; t < 4; ++t) {
    int c = lane + 32 * t;
    s += g1o[(size_t)row * CC + c] * w2[c];
  }
#pragma unroll
  for (int off = 16; off > 0; off >>= 1) s += __shfl_xor(s, off, 32);
  float gate = 1.0f / (1.0f + __expf(-(s + b2[0])));
  gate = fminf(fmaxf(gate, 0.1f), 1.0f);
  if (lane < 3) {
    float rm = 0.f;
    for (int h = 0; h < HH; ++h) rm += s_r[((size_t)row * HH + h) * 3 + lane];
    rm *= 0.25f;
    float ri = r[row * 3 + lane];
    r[row * 3 + lane] = ri + gate * (ri - rm);
  }
}

// ---------------------------------------------------------------------------
// MoE gating: softmax over ne logits, top-k, renormalize -> wfull[N,4].
// ---------------------------------------------------------------------------
__global__ void gating_kernel(const float* __restrict__ logits,
                              float* __restrict__ wfull, int ne, int kk) {
  int n = blockIdx.x * blockDim.x + threadIdx.x;
  if (n >= NN) return;
  float p[4] = {0.f, 0.f, 0.f, 0.f};
  float mx = -1e30f;
  for (int e = 0; e < ne; ++e) { p[e] = logits[n * ne + e]; mx = fmaxf(mx, p[e]); }
  float s = 0.f;
  for (int e = 0; e < ne; ++e) { p[e] = __expf(p[e] - mx); s += p[e]; }
  float invs = 1.0f / s;
  for (int e = 0; e < ne; ++e) p[e] *= invs;
  bool chosen[4] = {false, false, false, false};
  float csum = 0.f;
  for (int t = 0; t < kk; ++t) {
    int arg = -1; float best = -1.f;
    for (int e = 0; e < ne; ++e)
      if (!chosen[e] && p[e] > best) { best = p[e]; arg = e; }
    chosen[arg] = true; csum += p[arg];
  }
  float invc = 1.0f / csum;
  for (int e = 0; e < 4; ++e)
    wfull[n * 4 + e] = (e < ne && chosen[e]) ? p[e] * invc : 0.f;
}

// ---------------------------------------------------------------------------
// MoE combine: out[n,c] (=|+=) wfull[n,e] * x[n,c]
// ---------------------------------------------------------------------------
__global__ void combine_kernel(const float* __restrict__ xin,
                               const float* __restrict__ wfull,
                               float* __restrict__ out, int e, int first) {
  int idx = blockIdx.x * blockDim.x + threadIdx.x;   // N*C
  int n = idx >> 7;
  float val = wfull[n * 4 + e] * xin[idx];
  out[idx] = first ? val : out[idx] + val;
}

// ===========================================================================
// Host side
// ===========================================================================
struct LinP { const float* w; const float* b; };
struct LnP  { const float* g; const float* b; };
struct BlockP { LinP q, k, v, o, d1, d2, g1, g2, f1, f2; LnP ln1, ln2; };
struct ExpertP { LinP p1, p2; BlockP blk[2]; LnP ln; };
struct TaskP { int ne, tk; ExpertP exp[4]; LinP gl1, gl2, mp; };

extern "C" void kernel_launch(void* const* d_in, const int* in_sizes, int n_in,
                              void* d_out, int out_size, void* d_ws, size_t ws_size,
                              hipStream_t stream) {
  (void)in_sizes; (void)n_in; (void)out_size; (void)ws_size;

  const float* x     = (const float*)d_in[0];
  const int*   batch = (const int*)d_in[1];

  // --- walk flattened params in setup_inputs() dict insertion order ---
  int ii = 2;
  auto getLin = [&](LinP& l) { l.w = (const float*)d_in[ii++]; l.b = (const float*)d_in[ii++]; };
  auto getLn  = [&](LnP& l)  { l.g = (const float*)d_in[ii++]; l.b = (const float*)d_in[ii++]; };
  auto getBlock = [&](BlockP& b) {
    getLin(b.q); getLin(b.k); getLin(b.v); getLin(b.o);
    getLin(b.d1); getLin(b.d2); getLin(b.g1); getLin(b.g2);
    getLin(b.f1); getLin(b.f2); getLn(b.ln1); getLn(b.ln2);
  };
  auto getExpert = [&](ExpertP& e) {
    getLin(e.p1); getLin(e.p2); getBlock(e.blk[0]); getBlock(e.blk[1]); getLn(e.ln);
  };

  TaskP tasks[4];
  const int necfg[4] = {2, 2, 4, 1};   // energy, charge, geom, elem
  const int tkcfg[4] = {1, 1, 2, 1};
  for (int t = 0; t < 4; ++t) {
    tasks[t].ne = necfg[t]; tasks[t].tk = tkcfg[t];
    for (int e = 0; e < necfg[t]; ++e) getExpert(tasks[t].exp[e]);
    getLin(tasks[t].gl1); getLin(tasks[t].gl2); getLin(tasks[t].mp);
  }
  LinP he1, he2, hc1, hc2, hm1, hm2, cf1, cf2;
  getLin(he1); getLin(he2);           // head_energy l1, l2
  getLin(hc1); getLin(hc2);           // head_charge l1, l2
  getLin(hm1); getLin(hm2);           // head_elem l1, l2
  getLin(cf1); getLin(cf2);           // coord_fc1, coord_fc2

  // --- workspace carve (~4 MB of fp32) ---
  float* ws = (float*)d_ws;
  size_t off = 0;
  auto wsalloc = [&](size_t n) { float* p = ws + off; off += n; return p; };
  float* taskout[4];
  for (int t = 0; t < 4; ++t) taskout[t] = wsalloc((size_t)NN * CC);
  float* hbuf  = wsalloc((size_t)NN * CC);
  float* ghid  = wsalloc((size_t)NN * CC);
  float* glog  = wsalloc((size_t)NN * 4);
  float* wfull = wsalloc((size_t)NN * 4);
  float* ex    = wsalloc((size_t)NN * CC);      // expert state x
  float* exh   = wsalloc((size_t)NN * CC);      // LN output
  float* qb    = wsalloc((size_t)NN * CC);
  float* kb    = wsalloc((size_t)NN * CC);
  float* vb    = wsalloc((size_t)NN * CC);
  float* ctxb  = wsalloc((size_t)NN * CC);
  float* f1b   = wsalloc((size_t)NN * 2 * CC);
  float* g1b   = wsalloc((size_t)NN * CC);
  float* rb    = wsalloc((size_t)NN * 3);
  float* srb   = wsalloc((size_t)NN * HH * 3);
  float* headh = wsalloc((size_t)NN * CC);

  // --- launch helpers ---
  auto gemm = [&](const float* A, const LinP& L, const float* resid, float* out,
                  int K, int Dout, int act) {
    dim3 grid(NN / 64, Dout / 64), blk(32, 4);
    if (K == 128)
      gemm_wmma_kernel<128><<<grid, blk, 0, stream>>>(A, L.w, L.b, resid, out, Dout, act);
    else
      gemm_wmma_kernel<256><<<grid, blk, 0, stream>>>(A, L.w, L.b, resid, out, Dout, act);
  };
  auto lsmall = [&](const float* A, const LinP& L, float* out, int K, int Dout, int act) {
    int tot = NN * Dout;
    linear_small_kernel<<<(tot + 255) / 256, 256, 0, stream>>>(A, L.w, L.b, out, K, Dout, act, NN);
  };
  auto lnorm = [&](const float* xin, const LnP& L, float* out) {
    dim3 blk(32, 8);
    layernorm_kernel<<<NN / 8, blk, 0, stream>>>(xin, L.g, L.b, out);
  };

  auto run_block = [&](const BlockP& B) {
    lnorm(ex, B.ln1, exh);
    gemm(exh, B.q, nullptr, qb, CC, CC, 0);
    gemm(exh, B.k, nullptr, kb, CC, CC, 0);
    gemm(exh, B.v, nullptr, vb, CC, CC, 0);
    dim3 ablk(32, HH);
    attention_kernel<<<NN, ablk, 0, stream>>>(qb, kb, vb, rb, batch,
                                              B.d1.w, B.d1.b, B.d2.w, B.d2.b,
                                              ctxb, srb);
    gemm(ctxb, B.o, ex, ex, CC, CC, 0);             // x = x + o(ctx)
    lnorm(ex, B.ln2, exh);
    gemm(exh, B.f1, nullptr, f1b, CC, 2 * CC, 2);   // gelu(f1(.))
    gemm(f1b, B.f2, ex, ex, 2 * CC, CC, 0);         // x = x + f2(.)
    gemm(ex, B.g1, nullptr, g1b, CC, CC, 2);        // gelu(g1(x))
    dim3 rblk(32, 8);
    r_update_kernel<<<NN / 8, rblk, 0, stream>>>(g1b, B.g2.w, B.g2.b, srb, rb);
  };

  auto run_expert = [&](const ExpertP& E, const float* hin) {
    hipMemcpyAsync(ex, hin, (size_t)NN * CC * sizeof(float),
                   hipMemcpyDeviceToDevice, stream);
    gemm(hin, E.p1, nullptr, g1b, CC, CC, 2);       // gelu(p1(h))
    lsmall(g1b, E.p2, rb, CC, 3, 0);                // implicit coords r
    run_block(E.blk[0]);
    run_block(E.blk[1]);
    lnorm(ex, E.ln, ex);                            // final LN (row-local, in-place safe)
  };

  // --- per-task MoE ---
  for (int t = 0; t < 4; ++t) {
    TaskP& T = tasks[t];
    gemm(x, T.mp, nullptr, hbuf, CC, CC, 1);        // h = relu(map(x))
    gemm(hbuf, T.gl1, nullptr, ghid, CC, CC, 1);    // relu(gate.l1(h))
    lsmall(ghid, T.gl2, glog, CC, T.ne, 0);         // gate logits
    gating_kernel<<<(NN + 255) / 256, 256, 0, stream>>>(glog, wfull, T.ne, T.tk);
    for (int e = 0; e < T.ne; ++e) {
      run_expert(T.exp[e], hbuf);
      combine_kernel<<<(NN * CC) / 256, 256, 0, stream>>>(ex, wfull, taskout[t], e, e == 0);
    }
  }

  // --- heads into d_out: energy(512), charge(512), coords(1536), elem(48640) ---
  float* out = (float*)d_out;
  gemm(taskout[0], he1, nullptr, headh, CC, CC, 1);
  lsmall(headh, he2, out + 0, CC, 1, 0);
  gemm(taskout[1], hc1, nullptr, headh, CC, CC, 1);
  lsmall(headh, hc2, out + NN, CC, 1, 0);
  gemm(taskout[2], cf1, nullptr, headh, CC, CC, 1);
  lsmall(headh, cf2, out + 2 * NN, CC, 3, 0);
  gemm(taskout[3], hm1, nullptr, headh, CC, CC, 1);
  lsmall(headh, hm2, out + 5 * NN, CC, NELEM, 0);
}
